// SelfAttentionBlock_5411658793128
// MI455X (gfx1250) — compile-verified, matrix operand order checked by reading, not run on hardware
//
#include <hip/hip_runtime.h>
#include <hip/hip_bf16.h>

// ---------------------------------------------------------------------------
// SelfAttentionBlock on MI455X (gfx1250): GN -> QKV -> flash-attn -> proj.
// Matmuls via v_wmma_f32_16x16x32_f16 (wave32); attention K/V chunks staged
// into LDS with the Tensor Data Mover (tensor_load_to_lds, TENSORcnt),
// double-buffered so DMA of chunk i+1 overlaps WMMA compute of chunk i.
// ---------------------------------------------------------------------------

typedef __attribute__((ext_vector_type(16))) _Float16 v16h;
typedef __attribute__((ext_vector_type(8)))  _Float16 v8h;
typedef __attribute__((ext_vector_type(8)))  float    v8f;
typedef __attribute__((ext_vector_type(4)))  unsigned int u32x4;
typedef __attribute__((ext_vector_type(4)))  int          i32x4;
typedef __attribute__((ext_vector_type(8)))  int          i32x8;

#define BATCH 8
#define CH    256
#define NTOK  4096          // 64*64 tokens
#define GRP   8
#define CPG   32            // channels per group
#define EPSV  1e-5f

// ---------------------------------------------------------------------------
// Tile loader matching the CDNA5 16-bit 16x32 A-matrix VGPR layout
// (lanes 0-15: row=lane, K 0..7 & 16..23; lanes 16-31: row=lane-16,
//  K 8..15 & 24..31). The B layout is the mirror with row<->col, so the same
// loader serves B tiles when lane (=column) indexes rows of a row-major
// buffer. Two 16B loads per lane -> global_load_b128 / ds_load_b128.
// ---------------------------------------------------------------------------
__device__ __forceinline__ v16h ld_tile(const _Float16* p0, int stride) {
  const int lane = threadIdx.x & 31;
  const _Float16* p = p0 + (size_t)(lane & 15) * stride + ((lane >> 4) << 3);
  union { v16h v; v8h h[2]; } u;
  u.h[0] = *(const v8h*)(p);
  u.h[1] = *(const v8h*)(p + 16);
  return u.v;
}

// ---------------------------------------------------------------------------
// Tensor Data Mover: 2-D tile (global -> LDS), data_size = 2 bytes.
// D# built per CDNA5 ISA ch.8: group0 = {count, lds_addr, global_addr, type=2},
// group1 = {workgroup_mask=0, data_size=1(2B), tensor dims, tile dims,
// dim0 stride}. Groups 2/3 zero (<=2-D tensor). Dims in 2-byte elements.
// ---------------------------------------------------------------------------
__device__ __forceinline__ void tdm_load_2d(unsigned lds_addr,
                                            unsigned long long gaddr,
                                            unsigned tile_d0, unsigned tile_d1,
                                            unsigned tensor_d0, unsigned tensor_d1,
                                            unsigned long long stride0) {
  u32x4 g0;
  g0.x = 1u;                                              // count=1 (valid)
  g0.y = lds_addr;                                        // [63:32] LDS byte addr
  g0.z = (unsigned)(gaddr & 0xFFFFFFFFu);                 // [95:64]
  g0.w = (unsigned)((gaddr >> 32) & 0x1FFFFFFu) | (2u << 30); // [120:96]+type=2
  i32x8 g1;
  g1[0] = (int)(1u << 16);                                // data_size=1 -> 2B
  g1[1] = (int)((tensor_d0 & 0xFFFFu) << 16);             // tensor_dim0[15:0]
  g1[2] = (int)(((tensor_d0 >> 16) & 0xFFFFu) | ((tensor_d1 & 0xFFFFu) << 16));
  g1[3] = (int)(((tensor_d1 >> 16) & 0xFFFFu) | ((tile_d0 & 0xFFFFu) << 16));
  g1[4] = (int)(tile_d1 & 0xFFFFu);                       // tile_dim1 (tile_dim2=0)
  g1[5] = (int)(stride0 & 0xFFFFFFFFull);                 // dim0_stride[31:0]
  g1[6] = (int)((stride0 >> 32) & 0xFFFFu);               // dim0_stride[47:32]
  g1[7] = 0;                                              // dim1_stride unused (2-D)
  i32x4 z4 = (i32x4)0;
#if defined(__clang_major__) && (__clang_major__ >= 23)
  i32x8 z8 = (i32x8)0;
  __builtin_amdgcn_tensor_load_to_lds(g0, g1, z4, z4, z8, 0);
#else
  __builtin_amdgcn_tensor_load_to_lds(g0, g1, z4, z4, 0);
#endif
}

// Generic LDS address -> raw LDS byte offset (ISA: LDS aperture uses addr[31:0]).
__device__ __forceinline__ unsigned lds_off(const void* p) {
  return (unsigned)(unsigned long long)p;
}

// ---------------------------------------------------------------------------
// Kernel 1: GroupNorm statistics. One block per (b,g); 32 channels * 4096
// spatial = 131072 contiguous floats per group.
// ---------------------------------------------------------------------------
__global__ __launch_bounds__(256) void gn_stats_kernel(
    const float* __restrict__ x, float* __restrict__ stats) {
  __shared__ float ss[256], sq[256];
  const int blk = blockIdx.x;  // b*GRP + g
  const float* p = x + (size_t)blk * (CPG * NTOK);
  float s = 0.f, q = 0.f;
  for (int i = threadIdx.x; i < CPG * NTOK; i += 256) {
    float v = p[i];
    s += v; q += v * v;
  }
  ss[threadIdx.x] = s; sq[threadIdx.x] = q;
  __syncthreads();
  for (int off = 128; off > 0; off >>= 1) {
    if ((int)threadIdx.x < off) {
      ss[threadIdx.x] += ss[threadIdx.x + off];
      sq[threadIdx.x] += sq[threadIdx.x + off];
    }
    __syncthreads();
  }
  if (threadIdx.x == 0) {
    const float inv = 1.0f / (float)(CPG * NTOK);
    float mean = ss[0] * inv;
    float var  = sq[0] * inv - mean * mean;
    stats[2 * blk]     = mean;
    stats[2 * blk + 1] = rsqrtf(var + EPSV);
  }
}

// ---------------------------------------------------------------------------
// Kernel 2: apply GN affine and transpose (B,C,N) -> token-major (B,N,C) f16
// via a 32x32 LDS tile (both global sides coalesced).
// ---------------------------------------------------------------------------
__global__ __launch_bounds__(256) void gn_apply_kernel(
    const float* __restrict__ x, const float* __restrict__ stats,
    const float* __restrict__ gsc, const float* __restrict__ gbi,
    _Float16* __restrict__ z) {
  __shared__ float t[32][33];
  const int tx = threadIdx.x, ty = threadIdx.y;
  const int n0 = blockIdx.x * 32, c0 = blockIdx.y * 32, b = blockIdx.z;
#pragma unroll
  for (int i = 0; i < 4; ++i) {
    const int c = c0 + ty + i * 8;
    const int g = c >> 5;
    const float mean = stats[(b * GRP + g) * 2];
    const float rstd = stats[(b * GRP + g) * 2 + 1];
    const float v = x[(size_t)(b * CH + c) * NTOK + n0 + tx];
    t[ty + i * 8][tx] = (v - mean) * rstd * gsc[c] + gbi[c];
  }
  __syncthreads();
#pragma unroll
  for (int i = 0; i < 4; ++i) {
    const int n = n0 + ty + i * 8;
    const int c = c0 + tx;
    z[(size_t)(b * NTOK + n) * CH + c] = (_Float16)t[tx][ty + i * 8];
  }
}

// ---------------------------------------------------------------------------
// Kernel 3: convert weights f32 -> f16 once (re-read by thousands of waves).
// ---------------------------------------------------------------------------
__global__ __launch_bounds__(256) void cvt_w_kernel(
    const float* __restrict__ qkv_w, const float* __restrict__ proj_w,
    _Float16* __restrict__ qwh, _Float16* __restrict__ pwh) {
  const int i = blockIdx.x * blockDim.x + threadIdx.x;
  if (i < 3 * CH * CH) qwh[i] = (_Float16)qkv_w[i];
  if (i < CH * CH)     pwh[i] = (_Float16)proj_w[i];
}

// ---------------------------------------------------------------------------
// Kernel 4: QKV GEMM, M=B*N=32768, K=256, Nout=768. One wave = 16x64 tile.
// Q pre-scaled by 1/sqrt(C)=1/16; V stored transposed (B,C,N) so the P*V
// B-matrix loads contiguously per lane in the attention kernel.
// ---------------------------------------------------------------------------
__global__ __launch_bounds__(256) void qkv_gemm_kernel(
    const _Float16* __restrict__ z, const _Float16* __restrict__ wh,
    const float* __restrict__ qkv_b,
    _Float16* __restrict__ q, _Float16* __restrict__ kk,
    _Float16* __restrict__ vt) {
  const int wave = (int)(blockIdx.x * blockDim.x + threadIdx.x) >> 5;
  const int mi = wave & 2047;           // 2048 M tiles
  const int si = wave >> 11;            // 12 strips of 64 output cols
  const int m0 = mi * 16;
  const int lane = threadIdx.x & 31;

  v8f acc[4] = {};
#pragma unroll
  for (int kc = 0; kc < 8; ++kc) {
    const v16h a = ld_tile(z + (size_t)m0 * CH + kc * 32, CH);
#pragma unroll
    for (int j = 0; j < 4; ++j) {
      const int d0 = si * 64 + j * 16;
      const v16h bt = ld_tile(wh + (size_t)d0 * CH + kc * 32, CH);
      acc[j] = __builtin_amdgcn_wmma_f32_16x16x32_f16(
          false, a, false, bt, (short)0, acc[j], false, false);
    }
  }
  const int col = lane & 15;
  const int mh  = (lane >> 4) * 8;
#pragma unroll
  for (int j = 0; j < 4; ++j) {
    const int d = si * 64 + j * 16 + col;
    const float bias = qkv_b[d];
#pragma unroll
    for (int r = 0; r < 8; ++r) {
      const int t = m0 + mh + r;            // flat token index (b*N + n)
      const float v = acc[j][r] + bias;
      if (d < CH) {                          // Q, scaled by 1/sqrt(C)
        q[(size_t)t * CH + d] = (_Float16)(v * 0.0625f);
      } else if (d < 2 * CH) {               // K, row-major (B,N,C)
        kk[(size_t)t * CH + (d - CH)] = (_Float16)v;
      } else {                               // V, transposed (B,C,N)
        const int b = t >> 12, n = t & (NTOK - 1);
        vt[((size_t)b * CH + (d - 2 * CH)) * NTOK + n] = (_Float16)v;
      }
    }
  }
}

// ---------------------------------------------------------------------------
// Kernel 5: flash attention. Block = 4 waves, all on the same batch; each
// wave owns 16 query rows. K/V chunks (32 keys) are DMA'd into LDS by the
// TDM (wave 0 issues, TENSORcnt-tracked, double-buffered) and shared by all
// 4 waves; compute of chunk i overlaps the DMA of chunk i+1. Row max/sum use
// width-16 xor shuffles (C/D layout: the 16 columns of a row live across
// lanes 0-15 / 16-31). Probs round-trip through LDS to convert D-layout ->
// A-layout for the P*V WMMAs.
// ---------------------------------------------------------------------------
__global__ __launch_bounds__(128) void attn_kernel(
    const _Float16* __restrict__ q, const _Float16* __restrict__ k,
    const _Float16* __restrict__ vt, _Float16* __restrict__ ao) {
  __shared__ __align__(16) _Float16 Kls[2][32 * CH];   // 2 x 16 KB (rows=keys)
  __shared__ __align__(16) _Float16 Vls[2][CH * 32];   // 2 x 16 KB (rows=channels)
  __shared__ __align__(16) _Float16 plds[4][16 * 32];  // per-wave prob staging
  const int warp = threadIdx.x >> 5;
  const int lane = threadIdx.x & 31;
  const int w  = blockIdx.x * 4 + warp;     // 0..2047
  const int b  = w >> 8;                    // 256 query tiles per batch
  const int q0 = (w & 255) * 16;
  const _Float16* Qb = q  + ((size_t)b * NTOK + q0) * CH;
  const _Float16* Kb = k  + (size_t)b * NTOK * CH;
  const _Float16* Vb = vt + (size_t)b * CH * NTOK;
  _Float16* lp = &plds[warp][0];

  v16h qa[8];                                // Q tile resident in VGPRs
#pragma unroll
  for (int kc = 0; kc < 8; ++kc) qa[kc] = ld_tile(Qb + kc * 32, CH);

  v8f acc[16] = {};                          // 16x256 f32 output accumulators
  float mrun[8], lrun[8];
#pragma unroll
  for (int r = 0; r < 8; ++r) { mrun[r] = -1e30f; lrun[r] = 0.f; }

  const int mh = (lane >> 4) * 8;
  const int NCHUNK = NTOK / 32;

  // Preload chunk 0 into buffer 0 (2 TDM ops: K 1-D contiguous, V 2-D strided)
  if (warp == 0) {
    tdm_load_2d(lds_off(&Kls[0][0]), (unsigned long long)(const void*)Kb,
                8192u, 1u, 8192u, 1u, 8192ull);
    tdm_load_2d(lds_off(&Vls[0][0]), (unsigned long long)(const void*)Vb,
                32u, (unsigned)CH, (unsigned)NTOK, (unsigned)CH,
                (unsigned long long)NTOK);
  }

  for (int ci = 0; ci < NCHUNK; ++ci) {
    const int cur = ci & 1;
    if (warp == 0) {
      if (ci + 1 < NCHUNK) {
        const int key1 = (ci + 1) * 32;
        tdm_load_2d(lds_off(&Kls[cur ^ 1][0]),
                    (unsigned long long)(const void*)(Kb + (size_t)key1 * CH),
                    8192u, 1u, 8192u, 1u, 8192ull);
        tdm_load_2d(lds_off(&Vls[cur ^ 1][0]),
                    (unsigned long long)(const void*)(Vb + key1),
                    32u, (unsigned)CH, (unsigned)NTOK, (unsigned)CH,
                    (unsigned long long)NTOK);
        __builtin_amdgcn_s_wait_tensorcnt(2);  // chunk ci landed (in-order)
      } else {
        __builtin_amdgcn_s_wait_tensorcnt(0);
      }
    }
    __syncthreads();                            // publish Kls/Vls[cur]

    const _Float16* Kc = &Kls[cur][0];
    const _Float16* Vc = &Vls[cur][0];

    v8f s0 = {}, s1 = {};
#pragma unroll
    for (int kc = 0; kc < 8; ++kc) {
      const v16h b0 = ld_tile(Kc + kc * 32, CH);             // keys 0..15
      const v16h b1 = ld_tile(Kc + 16 * CH + kc * 32, CH);   // keys 16..31
      s0 = __builtin_amdgcn_wmma_f32_16x16x32_f16(
          false, qa[kc], false, b0, (short)0, s0, false, false);
      s1 = __builtin_amdgcn_wmma_f32_16x16x32_f16(
          false, qa[kc], false, b1, (short)0, s1, false, false);
    }
    // online softmax update per row
#pragma unroll
    for (int r = 0; r < 8; ++r) {
      float v = fmaxf(s0[r], s1[r]);
#pragma unroll
      for (int msk = 8; msk >= 1; msk >>= 1)
        v = fmaxf(v, __shfl_xor(v, msk, 32));
      const float mn = fmaxf(mrun[r], v);
      const float alpha = __expf(mrun[r] - mn);
      mrun[r] = mn;
      const float p0 = __expf(s0[r] - mn);
      const float p1 = __expf(s1[r] - mn);
      float rs = p0 + p1;
#pragma unroll
      for (int msk = 8; msk >= 1; msk >>= 1)
        rs += __shfl_xor(rs, msk, 32);
      lrun[r] = lrun[r] * alpha + rs;
#pragma unroll
      for (int j = 0; j < 16; ++j) acc[j][r] *= alpha;
      const int row = (r + mh) * 32;
      lp[row + (lane & 15)]      = (_Float16)p0;
      lp[row + 16 + (lane & 15)] = (_Float16)p1;
    }
    asm volatile("s_wait_dscnt 0" ::: "memory");   // D-layout stores -> A-layout loads
    const v16h pa = ld_tile(lp, 32);               // probs as 16x32 A-matrix
#pragma unroll
    for (int j = 0; j < 16; ++j) {
      const v16h bv = ld_tile(Vc + j * 16 * 32, 32);
      acc[j] = __builtin_amdgcn_wmma_f32_16x16x32_f16(
          false, pa, false, bv, (short)0, acc[j], false, false);
    }
    __syncthreads();   // all waves done reading buf[cur]; free for chunk ci+2
  }

  // normalize and emit f16 token-major (B,N,C)
#pragma unroll
  for (int j = 0; j < 16; ++j) {
#pragma unroll
    for (int r = 0; r < 8; ++r) {
      const int t = b * NTOK + q0 + mh + r;
      ao[(size_t)t * CH + j * 16 + (lane & 15)] =
          (_Float16)(acc[j][r] / lrun[r]);
    }
  }
}

// ---------------------------------------------------------------------------
// Kernel 6: output projection (M=32768, K=256, N=256) + bias + residual,
// writing f32 in the original (B,C,H,W) layout.
// ---------------------------------------------------------------------------
__global__ __launch_bounds__(256) void proj_kernel(
    const _Float16* __restrict__ ao, const _Float16* __restrict__ wh,
    const float* __restrict__ proj_b, const float* __restrict__ x,
    float* __restrict__ out) {
  const int wave = (int)(blockIdx.x * blockDim.x + threadIdx.x) >> 5;
  const int mi = wave & 2047;
  const int si = wave >> 11;   // 4 strips of 64
  const int m0 = mi * 16;
  const int lane = threadIdx.x & 31;

  v8f acc[4] = {};
#pragma unroll
  for (int kc = 0; kc < 8; ++kc) {
    const v16h a = ld_tile(ao + (size_t)m0 * CH + kc * 32, CH);
#pragma unroll
    for (int j = 0; j < 4; ++j) {
      const v16h bt = ld_tile(wh + (size_t)(si * 64 + j * 16) * CH + kc * 32, CH);
      acc[j] = __builtin_amdgcn_wmma_f32_16x16x32_f16(
          false, a, false, bt, (short)0, acc[j], false, false);
    }
  }
  const int col = lane & 15, mh = (lane >> 4) * 8;
#pragma unroll
  for (int j = 0; j < 4; ++j) {
    const int c = si * 64 + j * 16 + col;
    const float bias = proj_b[c];
#pragma unroll
    for (int r = 0; r < 8; ++r) {
      const int t = m0 + mh + r;
      const int b = t >> 12, n = t & (NTOK - 1);
      const size_t idx = ((size_t)(b * CH + c)) * NTOK + n;
      out[idx] = x[idx] + acc[j][r] + bias;
    }
  }
}

// ---------------------------------------------------------------------------
// Host side
// ---------------------------------------------------------------------------
extern "C" void kernel_launch(void* const* d_in, const int* in_sizes, int n_in,
                              void* d_out, int out_size, void* d_ws, size_t ws_size,
                              hipStream_t stream) {
  (void)in_sizes; (void)n_in; (void)out_size; (void)ws_size;
  const float* x        = (const float*)d_in[0];
  const float* gn_scale = (const float*)d_in[1];
  const float* gn_bias  = (const float*)d_in[2];
  const float* qkv_w    = (const float*)d_in[3];
  const float* qkv_b    = (const float*)d_in[4];
  const float* proj_w   = (const float*)d_in[5];
  const float* proj_b   = (const float*)d_in[6];
  float* out = (float*)d_out;

  // Workspace layout (~84.5 MB):
  char* ws = (char*)d_ws;
  const size_t R = (size_t)BATCH * NTOK * CH * sizeof(_Float16);  // 16 MB
  float*    stats = (float*)ws;                           // 512 B
  _Float16* z     = (_Float16*)(ws + 1024);               // (B,N,C) f16
  _Float16* q     = (_Float16*)(ws + 1024 + 1 * R);       // (B,N,C) f16, pre-scaled
  _Float16* k     = (_Float16*)(ws + 1024 + 2 * R);       // (B,N,C) f16
  _Float16* vt    = (_Float16*)(ws + 1024 + 3 * R);       // (B,C,N) f16 (transposed)
  _Float16* ao    = (_Float16*)(ws + 1024 + 4 * R);       // (B,N,C) f16
  _Float16* qwh   = (_Float16*)(ws + 1024 + 5 * R);       // qkv_w f16
  _Float16* pwh   = (_Float16*)(ws + 1024 + 5 * R + (size_t)3 * CH * CH * 2);

  gn_stats_kernel<<<BATCH * GRP, 256, 0, stream>>>(x, stats);

  dim3 gb(NTOK / 32, CH / 32, BATCH), tb(32, 8);
  gn_apply_kernel<<<gb, tb, 0, stream>>>(x, stats, gn_scale, gn_bias, z);

  cvt_w_kernel<<<(3 * CH * CH + 255) / 256, 256, 0, stream>>>(qkv_w, proj_w, qwh, pwh);

  // 2048 M-tiles * 12 strips = 24576 waves -> 3072 blocks of 8 waves
  qkv_gemm_kernel<<<3072, 256, 0, stream>>>(z, qwh, qkv_b, q, k, vt);

  // 8*256 query tiles = 2048 waves -> 512 blocks of 4 waves
  attn_kernel<<<512, 128, 0, stream>>>(q, k, vt, ao);

  // 2048 M-tiles * 4 strips = 8192 waves -> 1024 blocks of 8 waves
  proj_kernel<<<1024, 256, 0, stream>>>(ao, pwh, proj_b, x, out);
}
